// LGN_41274635714964
// MI455X (gfx1250) — matrix operand order repaired and checked by reading, not compiled, not run on hardware
//
#include <hip/hip_runtime.h>

#define IMG 224
#define PADV 6            // KS//2 * MAXSC = 3*2
#define XPD 236           // IMG + 2*PADV
#define BN 16
#define ICN 64
#define OCN 64
#define POOL 112

typedef __attribute__((ext_vector_type(16))) __bf16        v16bf;
typedef __attribute__((ext_vector_type(8)))  float         v8f;
typedef __attribute__((ext_vector_type(8)))  unsigned int  v8u;

union V16 { v16bf bf; unsigned short u[16]; v8u v; };

__device__ __forceinline__ unsigned short f2bf(float f) {
  __bf16 h = (__bf16)f;                     // native v_cvt_pk_bf16_f32 path (RNE)
  unsigned short u;
  __builtin_memcpy(&u, &h, 2);
  return u;
}
__device__ __forceinline__ float bf2f(unsigned short h) {
  return __uint_as_float(((unsigned)h) << 16);
}

// ---------------------------------------------------------------------------
// Zero-fill (xp pad ring must be 0; d_ws is poisoned 0xAA)
// ---------------------------------------------------------------------------
__global__ __launch_bounds__(256) void k_zero(uint4* __restrict__ p, int n) {
  int i = blockIdx.x * blockDim.x + threadIdx.x;
  int stride = gridDim.x * blockDim.x;
  uint4 z = {0u, 0u, 0u, 0u};
  for (; i < n; i += stride) p[i] = z;
}

// ---------------------------------------------------------------------------
// Pre-convert + swizzle w_conv (64,64,7,7) f32 into bf16 WMMA-A fragments:
//   wprep[((kc*4 + m)*32 + lane)*16 + e]
// so each lane's 16-element A fragment is 32 contiguous bytes (2x b128 loads).
// K ordering: K = tap*64 + ic  (chunk kc: tap = kc/2, ic base = (kc&1)*32).
// ---------------------------------------------------------------------------
__global__ __launch_bounds__(256) void k_wprep(const float* __restrict__ wconv,
                                               unsigned short* __restrict__ wprep) {
  int t = blockIdx.x * 256 + threadIdx.x;   // total = 98*4*32*16 = 200704
  if (t >= 98 * 4 * 32 * 16) return;
  int e    = t & 15;
  int lane = (t >> 4) & 31;
  int m    = (t >> 9) & 3;
  int kc   = t >> 11;
  int hi   = lane >> 4;
  int lp   = lane & 15;
  int oc   = m * 16 + lp;
  int icb  = (kc & 1) << 5;
  int tap  = kc >> 1;
  int lk   = hi * 16 + e;   // A 16x32 bf16 K layout (contiguous: k = 16*hi + e)
  int ic   = icb + lk;
  wprep[t] = f2bf(wconv[(oc * ICN + ic) * 49 + tap]);
}

// ---------------------------------------------------------------------------
// Kernel 1: fused concat(f,b) -> 1x1 conv (GEMM, K=128) -> padded bf16 xp.
// Per wave: one 16-pixel tile x all 4 OC tiles; 4 K-chunks x 4 WMMAs.
// w_fb staged to LDS as bf16 (A fragments become ds_load_b128 pairs).
// ---------------------------------------------------------------------------
__global__ __launch_bounds__(256) void k_conv1x1(const float* __restrict__ f,
                                                 const float* __restrict__ bsrc,
                                                 const float* __restrict__ wfb,
                                                 unsigned short* __restrict__ xp) {
  __shared__ unsigned short wsh[OCN * 128];
  for (int i = threadIdx.x; i < OCN * 128; i += 256) wsh[i] = f2bf(wfb[i]);
  __syncthreads();

  int lane = threadIdx.x & 31;
  int wave = threadIdx.x >> 5;
  int lp   = lane & 15;
  int hi   = (lane >> 4) & 1;

  int tile = blockIdx.x * 8 + wave;        // 50176 tiles exactly, no guard
  int p    = tile * 16 + lp;               // pixel id; tiles never straddle batch
  int b    = p / (IMG * IMG);
  int rem  = p - b * (IMG * IMG);
  int h    = rem / IMG;
  int w    = rem - h * IMG;

  const float* fp = f    + (b * ICN) * (IMG * IMG) + h * IMG + w;
  const float* bp = bsrc + (b * ICN) * (IMG * IMG) + h * IMG + w;

  v8f acc[4] = {};
  for (int kc = 0; kc < 4; ++kc) {
    V16 Bv;                                // B 32x16 bf16: lane-half K {0..15}/{16..31}
    int kb = kc * 32 + hi * 16;
#pragma unroll
    for (int e = 0; e < 16; ++e) {
      int ic = kb + e;
      float v = (ic < ICN) ? fp[ic * (IMG * IMG)] : bp[(ic - ICN) * (IMG * IMG)];
      Bv.u[e] = f2bf(v);
    }
#pragma unroll
    for (int m = 0; m < 4; ++m) {
      V16 Av;                              // A 16x32 bf16: contiguous K (k = 16*hi + e)
      int oc = m * 16 + lp;
#pragma unroll
      for (int e = 0; e < 16; ++e) {
        int lk = hi * 16 + e;
        Av.u[e] = wsh[oc * 128 + kc * 32 + lk];
      }
      acc[m] = __builtin_amdgcn_wmma_f32_16x16x32_bf16(
          false, Av.bf, false, Bv.bf, (short)0, acc[m], false, false);
    }
  }

  // D layout: VGPR r -> row r + 8*hi, col = lane&15
  unsigned short* op = xp + (b * OCN) * (XPD * XPD) + (PADV + h) * XPD + (PADV + w);
#pragma unroll
  for (int m = 0; m < 4; ++m)
#pragma unroll
    for (int r = 0; r < 8; ++r)
      op[(m * 16 + r + hi * 8) * (XPD * XPD)] = f2bf(acc[m][r]);
}

// ---------------------------------------------------------------------------
// Kernel 2: antialiased linear resize of window xp[ti:ti+hp, li:li+wp] -> ds.
// Triangle filter with support = scale (>=1), separable normalization.
// ds is stored PIXEL-MAJOR (ic innermost): ds[(b*area + oy*dw + ox)*64 + ic]
// so k_conv7's B fragments are contiguous 32B (2x global_load_b128).
// ---------------------------------------------------------------------------
__global__ __launch_bounds__(256) void k_resize(const unsigned short* __restrict__ xp,
                                                unsigned short* __restrict__ ds,
                                                int ti, int li, int hp, int wp,
                                                int dh, int dw) {
  int idx = blockIdx.x * 256 + threadIdx.x;
  int area = dh * dw;
  int total = BN * ICN * area;
  if (idx >= total) return;
  int ox = idx % dw;
  int oy = (idx / dw) % dh;
  int ch = idx / area;                      // b*64 + ic
  int ic = ch & 63;
  int b  = ch >> 6;

  float sy = (float)hp / (float)dh;
  float sx = (float)wp / (float)dw;
  float cy = (oy + 0.5f) * sy - 0.5f;
  float cx = (ox + 0.5f) * sx - 0.5f;
  int y0 = (int)ceilf(cy - sy);  if (y0 < 0) y0 = 0;
  int y1 = (int)floorf(cy + sy); if (y1 > hp - 1) y1 = hp - 1;
  int x0 = (int)ceilf(cx - sx);  if (x0 < 0) x0 = 0;
  int x1 = (int)floorf(cx + sx); if (x1 > wp - 1) x1 = wp - 1;

  const unsigned short* src = xp + ch * (XPD * XPD);
  float av = 0.f, wsum = 0.f;
  for (int iy = y0; iy <= y1; ++iy) {
    float wy = 1.f - fabsf((float)iy - cy) / sy; if (wy < 0.f) wy = 0.f;
    for (int ix = x0; ix <= x1; ++ix) {
      float wx = 1.f - fabsf((float)ix - cx) / sx; if (wx < 0.f) wx = 0.f;
      float ww = wy * wx;
      av += ww * bf2f(src[(ti + iy) * XPD + (li + ix)]);
      wsum += ww;
    }
  }
  ds[(b * area + oy * dw + ox) * ICN + ic] = f2bf(av / wsum);
}

// ---------------------------------------------------------------------------
// Kernel 3: 7x7 conv (pad 3) as implicit GEMM. K = tap*64 + ic => each 32-chunk
// has a fixed tap (one bounds check per chunk). 98 chunks x 4 WMMAs per tile.
// A: pre-swizzled wprep fragments; B: contiguous 32B pixel-major ds fragments.
// ---------------------------------------------------------------------------
__global__ __launch_bounds__(256) void k_conv7(const unsigned short* __restrict__ ds,
                                               const unsigned short* __restrict__ wprep,
                                               unsigned short* __restrict__ cw,
                                               int dh, int dw) {
  int lane = threadIdx.x & 31;
  int wave = threadIdx.x >> 5;
  int area = dh * dw;
  int total = BN * area;
  int tiles = (total + 15) >> 4;
  int tile = blockIdx.x * 8 + wave;
  if (tile >= tiles) return;                // uniform per wave: EXEC stays all-ones

  int lp = lane & 15;
  int hi = (lane >> 4) & 1;
  int p = tile * 16 + lp;
  int valid = p < total;
  if (!valid) p = total - 1;
  int b   = p / area;
  int rem = p - b * area;
  int y   = rem / dw;
  int x   = rem - y * dw;

  const unsigned short* wlane = wprep + lane * 16;   // + (kc*4+m)*512
  const unsigned short* dspix = ds + (size_t)b * area * ICN;
  v8u bz = {};

  v8f acc[4] = {};
  for (int kc = 0; kc < 98; ++kc) {
    int tap = kc >> 1;
    int icb = (kc & 1) << 5;
    int ky = tap / 7;
    int kx = tap - 7 * ky;
    int iy = y + ky - 3;
    int ix = x + kx - 3;
    int inb = (iy >= 0) & (iy < dh) & (ix >= 0) & (ix < dw);

    // prefetch next chunk's pre-swizzled weight block (perfectly predictable)
    __builtin_prefetch(wlane + (kc + 1) * 2048, 0, 3);

    V16 Bv;                                 // 16 consecutive ic = 32 contiguous bytes
    const v8u* dp = (const v8u*)(dspix + (size_t)(iy * dw + ix) * ICN + icb + hi * 16);
    Bv.v = inb ? *dp : bz;

#pragma unroll
    for (int m = 0; m < 4; ++m) {
      V16 Av;
      Av.v = *(const v8u*)(wlane + (kc * 4 + m) * 512);  // 32B contiguous
      acc[m] = __builtin_amdgcn_wmma_f32_16x16x32_bf16(
          false, Av.bf, false, Bv.bf, (short)0, acc[m], false, false);
    }
  }

  if (valid) {
    unsigned short* op = cw + (b * OCN) * area + y * dw + x;
#pragma unroll
    for (int m = 0; m < 4; ++m)
#pragma unroll
      for (int r = 0; r < 8; ++r)
        op[(m * 16 + r + hi * 8) * area] = f2bf(acc[m][r]);
  }
}

// ---------------------------------------------------------------------------
// Kernel 4: fused nearest-upsample (idx = i*dh//oh) + ReLU + 2x2 maxpool/argmax.
// Window out regions are 16-aligned, so each pool cell is inside one window.
// ---------------------------------------------------------------------------
__global__ __launch_bounds__(256) void k_pool(const unsigned short* __restrict__ cw,
                                              float* __restrict__ vals,
                                              int* __restrict__ idxo,
                                              int to, int lo, int oh, int ow,
                                              int dh, int dw) {
  int ph = oh >> 1, pw = ow >> 1;
  int total = BN * OCN * ph * pw;
  int t = blockIdx.x * 256 + threadIdx.x;
  if (t >= total) return;
  int pc = t % pw;
  int pr = (t / pw) % ph;
  int ch = t / (ph * pw);                   // b*64 + oc
  int area = dh * dw;
  const unsigned short* src = cw + ch * area;

  float best = -1.0f;
  int bidx = 0;
#pragma unroll
  for (int a = 0; a < 4; ++a) {             // arg order matches jnp.argmax
    int dy = a >> 1, dx = a & 1;
    int r = 2 * pr + dy, c = 2 * pc + dx;
    int cy = (r * dh) / oh, cx = (c * dw) / ow;
    float v = bf2f(src[cy * dw + cx]);
    if (v < 0.f) v = 0.f;                   // ReLU
    if (v > best) { best = v; bidx = (to + r) * IMG + (lo + c); }
  }
  int o = ch * (POOL * POOL) + (to / 2 + pr) * POOL + (lo / 2 + pc);
  vals[o] = best;
  idxo[o] = bidx;
}

// ---------------------------------------------------------------------------
extern "C" void kernel_launch(void* const* d_in, const int* in_sizes, int n_in,
                              void* d_out, int out_size, void* d_ws, size_t ws_size,
                              hipStream_t stream) {
  const float* f     = (const float*)d_in[0];
  const float* b     = (const float*)d_in[1];
  const float* wfb   = (const float*)d_in[2];
  const float* wconv = (const float*)d_in[3];

  // Workspace layout (bf16 stored as ushort):
  //   xp:    (16,64,236,236) = 57,032,704 elems (114 MB) -- fits MI455X 192MB L2
  //   ds:    (16,<=2048,64)  =  2,097,152 elems (4 MB)   -- pixel-major, max area 1540
  //   cw:    (16,64,<=2048)  =  2,097,152 elems (4 MB)   -- channel-major
  //   wprep: 98*4*32*16      =    200,704 elems (400 KB) -- swizzled A fragments
  unsigned short* xp = (unsigned short*)d_ws;
  size_t xp_elems = (size_t)BN * OCN * XPD * XPD;
  unsigned short* ds = xp + xp_elems;
  size_t slot = (size_t)BN * OCN * 2048;
  unsigned short* cw = ds + slot;
  unsigned short* wprep = cw + slot;

  float* vals = (float*)d_out;
  int*   idxo = ((int*)d_out) + (size_t)BN * OCN * POOL * POOL;

  int zero_n = (int)((xp_elems * 2) / 16);              // uint4 stores
  k_zero<<<dim3(4096), dim3(256), 0, stream>>>((uint4*)xp, zero_n);

  k_wprep<<<dim3((98 * 4 * 32 * 16 + 255) / 256), dim3(256), 0, stream>>>(wconv, wprep);

  k_conv1x1<<<dim3(50176 / 8), dim3(256), 0, stream>>>(f, b, wfb, xp);

  // Static window geometry (mirrors _geometry(): borders=16*s, scales=2-s/6)
  for (int s = 0; s < 7; ++s) {
    float scale = (float)(2.0 - (double)s * (1.0 / 6.0));
    int a = 16 * s, bb2 = 16 * (s + 1), c = IMG - bb2, d = IMG - a;
    int win[4][4] = {{a, a, bb2, c}, {bb2, a, d, bb2}, {c, bb2, d, d}, {a, c, c, d}};
    for (int wi = 0; wi < 4; ++wi) {
      int ti = win[wi][0], li = win[wi][1], bi = win[wi][2], ri = win[wi][3];
      int hp = bi - ti + 2 * PADV, wp = ri - li + 2 * PADV;
      int dh = (int)((float)hp / scale);                // matches int(np.float32/scale)
      int dw = (int)((float)wp / scale);
      int oh = bi - ti, ow = ri - li;

      int rn = BN * ICN * dh * dw;
      k_resize<<<dim3((rn + 255) / 256), dim3(256), 0, stream>>>(
          xp, ds, ti, li, hp, wp, dh, dw);

      int tiles = (BN * dh * dw + 15) / 16;
      k_conv7<<<dim3((tiles + 7) / 8), dim3(256), 0, stream>>>(ds, wprep, cw, dh, dw);

      int pn = BN * OCN * (oh / 2) * (ow / 2);
      k_pool<<<dim3((pn + 255) / 256), dim3(256), 0, stream>>>(
          cw, vals, idxo, ti, li, oh, ow, dh, dw);
    }
  }
}
